// EncoderBlock_17145509446317
// MI455X (gfx1250) — matrix operand order repaired and checked by reading, not compiled, not run on hardware
//
#include <hip/hip_runtime.h>
#include <hip/hip_bf16.h>

#define D_MODEL 1024
#define D_FF    4096
#define NH      16
#define DKH     64
#define BATCH   2
#define SEQ     2048
#define NTOK    (BATCH*SEQ)   // 4096

typedef __attribute__((ext_vector_type(16))) __bf16 v16bf;
typedef __attribute__((ext_vector_type(8)))  __bf16 v8bf;
typedef __attribute__((ext_vector_type(8)))  float  v8f;

__device__ __forceinline__ v16bf pack2(const __bf16* lo, const __bf16* hi) {
  v8bf a = *(const v8bf*)lo;
  v8bf b = *(const v8bf*)hi;
  v16bf r;
#pragma unroll
  for (int i = 0; i < 8; ++i) { r[i] = a[i]; r[i + 8] = b[i]; }
  return r;
}

// CDNA5 async global->LDS copy (16B per lane), tracked by ASYNCcnt.
// lds_addr = low 32 bits of generic LDS pointer (HW LDS byte address).
__device__ __forceinline__ void async_ld16(unsigned lds_addr, const __bf16* g) {
  asm volatile("global_load_async_to_lds_b128 %0, %1, off"
               :: "v"(lds_addr), "v"((unsigned long long)(uintptr_t)g)
               : "memory");
}
__device__ __forceinline__ void wait_async0() {
  asm volatile("s_wait_asynccnt 0x0" ::: "memory");
}

// ---------------------------------------------------------------------------
// f32 -> bf16 conversion (4 elements per thread)
// ---------------------------------------------------------------------------
__global__ __launch_bounds__(256) void f32_to_bf16_kernel(
    const float* __restrict__ in, __bf16* __restrict__ out, size_t n) {
  size_t i = ((size_t)blockIdx.x * 256 + threadIdx.x) * 4;
  if (i + 3 < n) {
#pragma unroll
    for (int j = 0; j < 4; ++j) out[i + j] = (__bf16)in[i + j];
  }
}

// ---------------------------------------------------------------------------
// Tiled bf16 GEMM: C[M,N] = act(A[M,K] @ B[K,N] + bias)
// block = 256 threads (8 waves); block tile 128x64; each wave 16x64; K-step 32.
// A tile staged via async global->LDS; B tile transposed through VGPRs.
// ---------------------------------------------------------------------------
template <int RELU>
__global__ __launch_bounds__(256) void gemm_bf16_kernel(
    const __bf16* __restrict__ A, const __bf16* __restrict__ Bm,
    const float* __restrict__ bias,
    float* __restrict__ Cf, __bf16* __restrict__ Cbf,
    int M, int N, int K) {
  __shared__ __bf16 lds_a[128][32];   // A tile, row-major (M x K-step), 8KB
  __shared__ __bf16 lds_bt[64][32];   // B tile transposed (N x K-step), 4KB

  const int t    = threadIdx.x;
  const int lane = t & 31;
  const int w    = t >> 5;
  const int row0 = blockIdx.y * 128;
  const int n0   = blockIdx.x * 64;
  const int m    = lane & 15;         // row within 16x16 (A) / col (B,C)
  const int half = lane >> 4;

  const unsigned lds_a_base = (unsigned)(uintptr_t)&lds_a[0][0];

  const v8f vz = {0.f, 0.f, 0.f, 0.f, 0.f, 0.f, 0.f, 0.f};
  v8f acc[4] = {vz, vz, vz, vz};

  for (int k0 = 0; k0 < K; k0 += 32) {
    // --- stage A tile via async copy: 128 rows x 32 cols, 2 x 16B per thread
    {
      const int ar = t >> 1, ch = t & 1;
      const __bf16* g = A + (size_t)(row0 + ar) * K + k0 + ch * 16;
      const unsigned l = lds_a_base + (unsigned)(ar * 64 + ch * 32);
      async_ld16(l, g);
      async_ld16(l + 16, g + 8);
    }
    // --- stage B tile transposed: read rows of B (coalesced), scatter to LDS
    {
      const int kk = t >> 3, ng = t & 7;
      v8bf bv = *(const v8bf*)(Bm + (size_t)(k0 + kk) * N + n0 + ng * 8);
#pragma unroll
      for (int i = 0; i < 8; ++i) lds_bt[ng * 8 + i][kk] = bv[i];
    }
    wait_async0();
    __syncthreads();

    // --- A fragment (16x32 layout: lanes 0-15 K=half*8.., VGPR4-7 K=16+half*8..)
    v16bf af = pack2(&lds_a[w * 16 + m][half * 8], &lds_a[w * 16 + m][16 + half * 8]);

#pragma unroll
    for (int nt = 0; nt < 4; ++nt) {
      // B fragment (32x16): column nt*16+m, K=half*8.. then K=16+half*8..
      v16bf bf = pack2(&lds_bt[nt * 16 + m][half * 8],
                       &lds_bt[nt * 16 + m][16 + half * 8]);
      acc[nt] = __builtin_amdgcn_wmma_f32_16x16x32_bf16(
          false, af, false, bf, (short)0, acc[nt], false, false);
    }
    __syncthreads();
  }

  // --- epilogue: bias (+ReLU), f32 and/or bf16 store
#pragma unroll
  for (int nt = 0; nt < 4; ++nt) {
    const int col = n0 + nt * 16 + m;
    const float bv = bias ? bias[col] : 0.f;
#pragma unroll
    for (int r = 0; r < 8; ++r) {
      const int grow = row0 + w * 16 + r + 8 * half;
      float v = acc[nt][r] + bv;
      if (RELU) v = fmaxf(v, 0.f);
      if (Cf)  Cf[(size_t)grow * N + col] = v;
      if (Cbf) Cbf[(size_t)grow * N + col] = (__bf16)v;
    }
  }
}

// ---------------------------------------------------------------------------
// V transpose: v[b*S+s][h*64+d] -> vt[((b*NH+h)*64+d)*SEQ + s]
// ---------------------------------------------------------------------------
__global__ __launch_bounds__(256) void transpose_v_kernel(
    const __bf16* __restrict__ V, __bf16* __restrict__ Vt) {
  size_t i = (size_t)blockIdx.x * 256 + threadIdx.x;  // over NTOK*D_MODEL
  if (i >= (size_t)NTOK * D_MODEL) return;
  const int d = (int)(i & 63);
  size_t tmp = i >> 6;
  const int h = (int)(tmp & 15); tmp >>= 4;
  const int s = (int)(tmp & (SEQ - 1));
  const int b = (int)(tmp >> 11);
  Vt[(((size_t)b * NH + h) * DKH + d) * SEQ + s] = V[i];
}

// ---------------------------------------------------------------------------
// Flash attention: one wave per 16-row Q tile of one (b,h); key blocks of 32.
// block = 128 threads (4 waves); grid = (SEQ/64, NH, BATCH)
// K (32x64) and V^T (64x32) tiles are async-staged into LDS once per key
// block and shared by all 4 waves.
// ---------------------------------------------------------------------------
__global__ __launch_bounds__(128) void attention_kernel(
    const __bf16* __restrict__ Q, const __bf16* __restrict__ Kb,
    const __bf16* __restrict__ Vt, const unsigned char* __restrict__ mask,
    __bf16* __restrict__ ctx) {
  __shared__ __bf16 lds_k[32][64];      // K tile  (key x d),   4KB
  __shared__ __bf16 lds_v[64][32];      // V^T tile (d x key),  4KB
  __shared__ __bf16 lds_p[4][16][32];   // per-wave P tile,     4KB

  const int t    = threadIdx.x;
  const int lane = t & 31;
  const int w    = t >> 5;
  const int h    = blockIdx.y;
  const int b    = blockIdx.z;
  const int q0   = blockIdx.x * 64 + w * 16;
  const int n    = lane & 15;
  const int half = lane >> 4;

  const unsigned lds_k_base = (unsigned)(uintptr_t)&lds_k[0][0];
  const unsigned lds_v_base = (unsigned)(uintptr_t)&lds_v[0][0];

  // Q fragments for d = 0..31 and 32..63 (A-matrix 16x32 layout)
  const __bf16* qrow = Q + ((size_t)(b * SEQ + q0 + n)) * D_MODEL + h * DKH;
  const v16bf qa0 = pack2(qrow + half * 8,       qrow + 16 + half * 8);
  const v16bf qa1 = pack2(qrow + 32 + half * 8,  qrow + 48 + half * 8);

  const v8f vz = {0.f, 0.f, 0.f, 0.f, 0.f, 0.f, 0.f, 0.f};
  v8f o[4] = {vz, vz, vz, vz};
  float mi[8], li[8];
#pragma unroll
  for (int r = 0; r < 8; ++r) { mi[r] = -INFINITY; li[r] = 0.f; }

  const __bf16* k_bh  = Kb + ((size_t)(b * SEQ)) * D_MODEL + h * DKH;
  const __bf16* vt_bh = Vt + ((size_t)(b * NH + h)) * DKH * SEQ;

  for (int key0 = 0; key0 < SEQ; key0 += 32) {
    // ---- async-stage K tile (32 rows x 64 d) and V^T tile (64 d x 32 keys)
#pragma unroll
    for (int j = 0; j < 2; ++j) {
      const int c = t * 2 + j;                        // 0..255, 16B chunks
      {  // K tile: row stride 128B, 8 chunks/row
        const int row = c >> 3, cc = c & 7;
        async_ld16(lds_k_base + (unsigned)(row * 128 + cc * 16),
                   k_bh + (size_t)(key0 + row) * D_MODEL + cc * 8);
      }
      {  // V^T tile: row stride 64B, 4 chunks/row
        const int row = c >> 2, cc = c & 3;
        async_ld16(lds_v_base + (unsigned)(row * 64 + cc * 16),
                   vt_bh + (size_t)row * SEQ + key0 + cc * 8);
      }
    }
    wait_async0();
    __syncthreads();

    // ---- scores tile 16x32 = Q (16x64) @ K^T, as 2 col-halves x 2 K-steps
    v8f s0 = vz, s1 = vz;
    {
      v16bf kf;
      kf = pack2(&lds_k[n][half * 8],            &lds_k[n][16 + half * 8]);
      s0 = __builtin_amdgcn_wmma_f32_16x16x32_bf16(false, qa0, false, kf, (short)0, s0, false, false);
      kf = pack2(&lds_k[n][32 + half * 8],       &lds_k[n][48 + half * 8]);
      s0 = __builtin_amdgcn_wmma_f32_16x16x32_bf16(false, qa1, false, kf, (short)0, s0, false, false);
      kf = pack2(&lds_k[16 + n][half * 8],       &lds_k[16 + n][16 + half * 8]);
      s1 = __builtin_amdgcn_wmma_f32_16x16x32_bf16(false, qa0, false, kf, (short)0, s1, false, false);
      kf = pack2(&lds_k[16 + n][32 + half * 8],  &lds_k[16 + n][48 + half * 8]);
      s1 = __builtin_amdgcn_wmma_f32_16x16x32_bf16(false, qa1, false, kf, (short)0, s1, false, false);
    }

    const bool msk0 = mask[(size_t)b * SEQ + key0 + n] != 0;
    const bool msk1 = mask[(size_t)b * SEQ + key0 + 16 + n] != 0;

    // ---- online softmax (row stats via 16-lane butterfly within each half)
#pragma unroll
    for (int r = 0; r < 8; ++r) {
      float a0 = msk0 ? -INFINITY : s0[r] * 0.125f;  // 1/sqrt(64)
      float a1 = msk1 ? -INFINITY : s1[r] * 0.125f;
      float rm = fmaxf(a0, a1);
#pragma unroll
      for (int off = 1; off < 16; off <<= 1) rm = fmaxf(rm, __shfl_xor(rm, off, 32));
      const float mnew = fmaxf(mi[r], rm);
      const float p0 = __expf(a0 - mnew);
      const float p1 = __expf(a1 - mnew);
      float rs = p0 + p1;
#pragma unroll
      for (int off = 1; off < 16; off <<= 1) rs += __shfl_xor(rs, off, 32);
      const float corr = __expf(mi[r] - mnew);
      li[r] = li[r] * corr + rs;
      mi[r] = mnew;
#pragma unroll
      for (int dt = 0; dt < 4; ++dt) o[dt][r] *= corr;
      const int mrow = r + 8 * half;
      lds_p[w][mrow][n]      = (__bf16)p0;
      lds_p[w][mrow][16 + n] = (__bf16)p1;
    }
    // per-wave LDS round-trip: wave-ordered DS + explicit CDNA5 wait
    asm volatile("s_wait_dscnt 0x0" ::: "memory");

    // ---- P fragment (A-layout 16x32) and PV accumulation over 4 d-tiles
    const v16bf pa = pack2(&lds_p[w][n][half * 8], &lds_p[w][n][16 + half * 8]);
#pragma unroll
    for (int dt = 0; dt < 4; ++dt) {
      v16bf vf = pack2(&lds_v[dt * 16 + n][half * 8],
                       &lds_v[dt * 16 + n][16 + half * 8]);
      o[dt] = __builtin_amdgcn_wmma_f32_16x16x32_bf16(
          false, pa, false, vf, (short)0, o[dt], false, false);
    }
    __syncthreads();   // protect K/V tiles before next stage
  }

  // ---- normalize and write ctx (bf16, [b][s][h*64+d])
#pragma unroll
  for (int dt = 0; dt < 4; ++dt) {
#pragma unroll
    for (int r = 0; r < 8; ++r) {
      const int mrow = r + 8 * half;
      const float val = o[dt][r] / li[r];
      ctx[((size_t)(b * SEQ + q0 + mrow)) * D_MODEL + h * DKH + dt * 16 + n] = (__bf16)val;
    }
  }
}

// ---------------------------------------------------------------------------
// Residual + LayerNorm (unbiased var, ddof=1); emits f32 and/or bf16
// one block (256 thr) per row of 1024
// ---------------------------------------------------------------------------
__global__ __launch_bounds__(256) void ln_kernel(
    const float* __restrict__ X, const float* __restrict__ R,
    const float* __restrict__ gamma, const float* __restrict__ beta,
    float* __restrict__ Yf, __bf16* __restrict__ Ybf) {
  __shared__ float sh_s[8], sh_ss[8];
  const int row = blockIdx.x;
  const int t = threadIdx.x, lane = t & 31, w = t >> 5;

  const float4 xv = *(const float4*)(X + (size_t)row * D_MODEL + t * 4);
  const float4 rv = *(const float4*)(R + (size_t)row * D_MODEL + t * 4);
  float v[4] = {xv.x + rv.x, xv.y + rv.y, xv.z + rv.z, xv.w + rv.w};

  float s = v[0] + v[1] + v[2] + v[3];
  float ss = v[0] * v[0] + v[1] * v[1] + v[2] * v[2] + v[3] * v[3];
#pragma unroll
  for (int off = 1; off < 32; off <<= 1) {
    s += __shfl_xor(s, off, 32);
    ss += __shfl_xor(ss, off, 32);
  }
  if (lane == 0) { sh_s[w] = s; sh_ss[w] = ss; }
  __syncthreads();
  float tot = 0.f, tots = 0.f;
#pragma unroll
  for (int i = 0; i < 8; ++i) { tot += sh_s[i]; tots += sh_ss[i]; }

  const float mean = tot / (float)D_MODEL;
  const float var  = (tots - (float)D_MODEL * mean * mean) / (float)(D_MODEL - 1);
  const float rstd = rsqrtf(var + 1e-5f);

#pragma unroll
  for (int i = 0; i < 4; ++i) {
    const int c = t * 4 + i;
    const float y = (v[i] - mean) * rstd * gamma[c] + beta[c];
    if (Yf)  Yf[(size_t)row * D_MODEL + c] = y;
    if (Ybf) Ybf[(size_t)row * D_MODEL + c] = (__bf16)y;
  }
}

// ---------------------------------------------------------------------------
// host-side orchestration
// ---------------------------------------------------------------------------
extern "C" void kernel_launch(void* const* d_in, const int* in_sizes, int n_in,
                              void* d_out, int out_size, void* d_ws, size_t ws_size,
                              hipStream_t stream) {
  const float* x            = (const float*)d_in[0];
  const unsigned char* mask = (const unsigned char*)d_in[1];
  const float* Wq = (const float*)d_in[2];  const float* bq = (const float*)d_in[3];
  const float* Wk = (const float*)d_in[4];  const float* bk = (const float*)d_in[5];
  const float* Wv = (const float*)d_in[6];  const float* bv = (const float*)d_in[7];
  const float* Wo = (const float*)d_in[8];  const float* bo = (const float*)d_in[9];
  const float* g1 = (const float*)d_in[10]; const float* be1 = (const float*)d_in[11];
  const float* W1 = (const float*)d_in[12]; const float* b1 = (const float*)d_in[13];
  const float* W2 = (const float*)d_in[14]; const float* b2 = (const float*)d_in[15];
  const float* g2 = (const float*)d_in[16]; const float* be2 = (const float*)d_in[17];
  float* out = (float*)d_out;

  char* ws = (char*)d_ws;
  auto alloc_bf = [&](size_t n) -> __bf16* {
    __bf16* p = (__bf16*)ws; ws += ((n * 2 + 255) / 256) * 256; return p;
  };
  auto alloc_f = [&](size_t n) -> float* {
    float* p = (float*)ws; ws += ((n * 4 + 255) / 256) * 256; return p;
  };

  const size_t ND = (size_t)NTOK * D_MODEL;
  __bf16* x_bf   = alloc_bf(ND);
  __bf16* wq_bf  = alloc_bf((size_t)D_MODEL * D_MODEL);
  __bf16* wk_bf  = alloc_bf((size_t)D_MODEL * D_MODEL);
  __bf16* wv_bf  = alloc_bf((size_t)D_MODEL * D_MODEL);
  __bf16* wo_bf  = alloc_bf((size_t)D_MODEL * D_MODEL);
  __bf16* w1_bf  = alloc_bf((size_t)D_MODEL * D_FF);
  __bf16* w2_bf  = alloc_bf((size_t)D_FF * D_MODEL);
  __bf16* q_bf   = alloc_bf(ND);
  __bf16* k_bf   = alloc_bf(ND);
  __bf16* v_bf   = alloc_bf(ND);
  __bf16* vt_bf  = alloc_bf(ND);
  __bf16* h_bf   = alloc_bf(ND);
  __bf16* ff1_bf = alloc_bf((size_t)NTOK * D_FF);
  float*  attn_f = alloc_f(ND);       // reused for ff2
  float*  h_f    = alloc_f(ND);
  __bf16* ctx_bf = v_bf;              // v_bf dead after transpose
  float*  ff2_f  = attn_f;            // attn_f dead after LN1

  auto cvt = [&](const float* src, __bf16* dst, size_t n) {
    f32_to_bf16_kernel<<<dim3((unsigned)((n / 4 + 255) / 256)), dim3(256), 0, stream>>>(src, dst, n);
  };
  cvt(x,  x_bf,  ND);
  cvt(Wq, wq_bf, (size_t)D_MODEL * D_MODEL);
  cvt(Wk, wk_bf, (size_t)D_MODEL * D_MODEL);
  cvt(Wv, wv_bf, (size_t)D_MODEL * D_MODEL);
  cvt(Wo, wo_bf, (size_t)D_MODEL * D_MODEL);
  cvt(W1, w1_bf, (size_t)D_MODEL * D_FF);
  cvt(W2, w2_bf, (size_t)D_FF * D_MODEL);

  const dim3 gblk(256);
  // QKV projections (M=4096, N=1024, K=1024) -> bf16
  gemm_bf16_kernel<0><<<dim3(D_MODEL / 64, NTOK / 128), gblk, 0, stream>>>(
      x_bf, wq_bf, bq, (float*)nullptr, q_bf, NTOK, D_MODEL, D_MODEL);
  gemm_bf16_kernel<0><<<dim3(D_MODEL / 64, NTOK / 128), gblk, 0, stream>>>(
      x_bf, wk_bf, bk, (float*)nullptr, k_bf, NTOK, D_MODEL, D_MODEL);
  gemm_bf16_kernel<0><<<dim3(D_MODEL / 64, NTOK / 128), gblk, 0, stream>>>(
      x_bf, wv_bf, bv, (float*)nullptr, v_bf, NTOK, D_MODEL, D_MODEL);

  transpose_v_kernel<<<dim3((unsigned)(ND / 256)), dim3(256), 0, stream>>>(v_bf, vt_bf);

  attention_kernel<<<dim3(SEQ / 64, NH, BATCH), dim3(128), 0, stream>>>(
      q_bf, k_bf, vt_bf, mask, ctx_bf);

  // output projection -> f32
  gemm_bf16_kernel<0><<<dim3(D_MODEL / 64, NTOK / 128), gblk, 0, stream>>>(
      ctx_bf, wo_bf, bo, attn_f, (__bf16*)nullptr, NTOK, D_MODEL, D_MODEL);

  // h = LN(attn_out + x)
  ln_kernel<<<dim3(NTOK), dim3(256), 0, stream>>>(attn_f, x, g1, be1, h_f, h_bf);

  // FFN: relu(h @ W1 + b1) -> bf16 ; @ W2 + b2 -> f32
  gemm_bf16_kernel<1><<<dim3(D_FF / 64, NTOK / 128), gblk, 0, stream>>>(
      h_bf, w1_bf, b1, (float*)nullptr, ff1_bf, NTOK, D_FF, D_MODEL);
  gemm_bf16_kernel<0><<<dim3(D_MODEL / 64, NTOK / 128), gblk, 0, stream>>>(
      ff1_bf, w2_bf, b2, ff2_f, (__bf16*)nullptr, NTOK, D_MODEL, D_FF);

  // out = LN(ff + h)
  ln_kernel<<<dim3(NTOK), dim3(256), 0, stream>>>(ff2_f, h_f, g2, be2, out, (__bf16*)nullptr);
}
